// SNNTorchSyn_91319594648103
// MI455X (gfx1250) — matrix operand order repaired and checked by reading, not compiled, not run on hardware
//
#include <hip/hip_runtime.h>

// ---------------- SNN constants (match reference) ----------------
#define ALPHA 0.8187307530779818f   // exp(-0.01/0.05)
#define BETA  0.9048374180359595f   // exp(-0.01/0.1)
#define THR   1.0f

// ---------------- vector types for WMMA ----------------
typedef __attribute__((ext_vector_type(16))) __bf16        v16bf;
typedef __attribute__((ext_vector_type(8)))  __bf16        v8bf;
typedef __attribute__((ext_vector_type(2)))  __bf16        v2bf;
typedef __attribute__((ext_vector_type(8)))  float         v8f;
typedef __attribute__((ext_vector_type(8)))  unsigned short v8us;

__device__ __forceinline__ unsigned pack2bf(float a, float b) {
    // one v_cvt_pk_bf16_f32 -> packed {lo=a, hi=b}
#if __has_builtin(__builtin_amdgcn_cvt_pk_bf16_f32)
    v2bf v = __builtin_amdgcn_cvt_pk_bf16_f32(a, b);
#else
    v2bf v = { (__bf16)a, (__bf16)b };
#endif
    return __builtin_bit_cast(unsigned, v);
}

__device__ __forceinline__ uint2 pack4bf(float4 f) {
    uint2 r;
    r.x = pack2bf(f.x, f.y);
    r.y = pack2bf(f.z, f.w);
    return r;
}

// ================= Kernel 1: w1 fp32 -> bf16 pack =================
__global__ __launch_bounds__(256) void cvt_w1_kernel(
    const float* __restrict__ w, unsigned short* __restrict__ o, int n4) {
    int i = blockIdx.x * 256 + threadIdx.x;
    if (i >= n4) return;
    ((uint2*)o)[i] = pack4bf(((const float4*)w)[i]);
}

// ================= Kernel 2: fc1 GEMM (bf16 WMMA) =================
// C[M,N] = A[M,K] * B[N,K]^T + b1,  M=25600, K=4096, N=1024
// Block tile 128x128, K-step 32, 8 waves: 2(M) x 4(N), each wave 64x32 = 4x2 WMMA tiles.
// Double-buffered LDS; B tile staged with global_load_async_to_lds_b128 (ASYNCcnt).
__global__ __launch_bounds__(256) void fc1_gemm_kernel(
    const float* __restrict__ x,              // [M, 4096] fp32 (converted in-flight)
    const unsigned short* __restrict__ w1bf,  // [1024, 4096] bf16 bits
    const float* __restrict__ b1,             // [1024]
    float* __restrict__ cur1)                 // [M, 1024] fp32
{
    constexpr int K = 4096;
    constexpr int BM = 128, BN = 128, BK = 32;
    constexpr int LDT = 40;  // padded LDS row (elements) -> conflict-free b128 reads

    __shared__ unsigned short sA[2][BM * LDT];
    __shared__ unsigned short sB[2][BN * LDT];

    const int tid  = threadIdx.x;
    const int lane = tid & 31;
    const int wid  = tid >> 5;
    const int wm   = wid >> 2;          // 0..1
    const int wn   = wid & 3;           // 0..3
    const int kh   = lane >> 4;         // lane half (0/1)
    const int l16  = lane & 15;
    const int tileM = blockIdx.y * BM;
    const int tileN = blockIdx.x * BN;

    // staging maps (per thread):
    // A: 4 float4 chunks; chunk = tid+256*i -> row = chunk/8, c4 = chunk%8
    // B: 2 x 16B chunks;  chunk = tid+256*i -> row = chunk/4, c8 = chunk%4
    const int arow0 = tid >> 3, ac4 = tid & 7;
    const int brow0 = tid >> 2, bc8 = tid & 3;

    v8f acc[4][2] = {};

    // ---- prologue: stage K-slab 0 into buffer 0 ----
    #pragma unroll
    for (int i = 0; i < 4; ++i) {
        int row = arow0 + 32 * i;
        float4 f = *(const float4*)(x + (size_t)(tileM + row) * K + ac4 * 4);
        *(uint2*)&sA[0][row * LDT + ac4 * 4] = pack4bf(f);
    }
    #pragma unroll
    for (int i = 0; i < 2; ++i) {
        int row = brow0 + 64 * i;
        unsigned lds = (unsigned)(uintptr_t)&sB[0][row * LDT + bc8 * 8];
        unsigned long long ga =
            (unsigned long long)(uintptr_t)(w1bf + (size_t)(tileN + row) * K + bc8 * 8);
        asm volatile("global_load_async_to_lds_b128 %0, %1, off"
                     :: "v"(lds), "v"(ga) : "memory");
    }
    asm volatile("s_wait_asynccnt 0x0" ::: "memory");
    __syncthreads();

    // ---- main loop: compute on buf, stage k0+BK into buf^1 ----
    for (int k0 = 0, buf = 0; k0 < K; k0 += BK, buf ^= 1) {
        const bool has_next = (k0 + BK) < K;
        const int k1 = k0 + BK;

        float4 areg[4];
        if (has_next) {
            // issue next-slab global ops before the WMMA burst
            #pragma unroll
            for (int i = 0; i < 4; ++i) {
                int row = arow0 + 32 * i;
                const float* gp = x + (size_t)(tileM + row) * K + k1 + ac4 * 4;
                areg[i] = *(const float4*)gp;
                __builtin_prefetch(gp + BK, 0, 3);  // k0+2*BK slab -> L2
            }
            #pragma unroll
            for (int i = 0; i < 2; ++i) {
                int row = brow0 + 64 * i;
                unsigned lds = (unsigned)(uintptr_t)&sB[buf ^ 1][row * LDT + bc8 * 8];
                unsigned long long ga =
                    (unsigned long long)(uintptr_t)(w1bf + (size_t)(tileN + row) * K + k1 + bc8 * 8);
                asm volatile("global_load_async_to_lds_b128 %0, %1, off"
                             :: "v"(lds), "v"(ga) : "memory");
            }
        }

        // ---- fragments per ISA 7.12.2 layouts ----
        const unsigned short* cA = sA[buf];
        const unsigned short* cB = sB[buf];
        v16bf afrag[4], bfrag[2];
        #pragma unroll
        for (int i = 0; i < 4; ++i) {
            int row = wm * 64 + i * 16 + l16;     // A: M = lane%16
            v8us lo = *(const v8us*)&cA[row * LDT + kh * 8];        // K = kh*8+0..7
            v8us hi = *(const v8us*)&cA[row * LDT + 16 + kh * 8];   // K = 16+kh*8+0..7
            afrag[i] = __builtin_shufflevector(
                __builtin_bit_cast(v8bf, lo), __builtin_bit_cast(v8bf, hi),
                0,1,2,3,4,5,6,7,8,9,10,11,12,13,14,15);
        }
        #pragma unroll
        for (int j = 0; j < 2; ++j) {
            int col = wn * 32 + j * 16 + l16;     // B: N = lane%16
            v8us lo = *(const v8us*)&cB[col * LDT + kh * 8];        // K = kh*8+0..7
            v8us hi = *(const v8us*)&cB[col * LDT + 16 + kh * 8];   // K = 16+kh*8+0..7
            bfrag[j] = __builtin_shufflevector(
                __builtin_bit_cast(v8bf, lo), __builtin_bit_cast(v8bf, hi),
                0,1,2,3,4,5,6,7,8,9,10,11,12,13,14,15);
        }
        #pragma unroll
        for (int i = 0; i < 4; ++i)
            #pragma unroll
            for (int j = 0; j < 2; ++j)
                acc[i][j] = __builtin_amdgcn_wmma_f32_16x16x32_bf16(
                    false, afrag[i], false, bfrag[j],
                    (short)0, acc[i][j], false, false);

        // ---- write next A slab (convert after compute to overlap loads) ----
        if (has_next) {
            #pragma unroll
            for (int i = 0; i < 4; ++i) {
                int row = arow0 + 32 * i;
                *(uint2*)&sA[buf ^ 1][row * LDT + ac4 * 4] = pack4bf(areg[i]);
            }
        }
        asm volatile("s_wait_asynccnt 0x0" ::: "memory");
        __syncthreads();
    }

    // ---- epilogue: C layout (VGPR v -> M = v + 8*(lane>=16)), fused bias ----
    #pragma unroll
    for (int j = 0; j < 2; ++j) {
        int col = tileN + wn * 32 + j * 16 + l16;
        float bb = b1[col];
        #pragma unroll
        for (int i = 0; i < 4; ++i) {
            int rbase = tileM + wm * 64 + i * 16 + (kh ? 8 : 0);
            #pragma unroll
            for (int v = 0; v < 8; ++v)
                cur1[(size_t)(rbase + v) * 1024 + col] = acc[i][j][v] + bb;
        }
    }
}

// ================= Kernel 3: temporal scan =================
__device__ __forceinline__ float wave_sum(float x) {
    // xor-tree over 32 lanes via ds_swizzle (group-of-32: imm = xor_mask<<10 | 0x1F)
    x += __builtin_bit_cast(float, __builtin_amdgcn_ds_swizzle(__builtin_bit_cast(int, x), 0x401F));
    x += __builtin_bit_cast(float, __builtin_amdgcn_ds_swizzle(__builtin_bit_cast(int, x), 0x201F));
    x += __builtin_bit_cast(float, __builtin_amdgcn_ds_swizzle(__builtin_bit_cast(int, x), 0x101F));
    x += __builtin_bit_cast(float, __builtin_amdgcn_ds_swizzle(__builtin_bit_cast(int, x), 0x081F));
    x += __builtin_bit_cast(float, __builtin_amdgcn_ds_swizzle(__builtin_bit_cast(int, x), 0x041F));
    return x;
}

__global__ __launch_bounds__(256) void snn_scan_kernel(
    const float* __restrict__ cur1,  // [B*T, 1024], row m = b*T + t
    const float* __restrict__ w2,    // [2, 1024]
    const float* __restrict__ b2,    // [2]
    float* __restrict__ out)         // [mem2: B,T,2][spk2: B,T,2]
{
    constexpr int T = 200, NH = 1024, B = 128;
    const int b = blockIdx.x, tid = threadIdx.x;
    const int lane = tid & 31, wid = tid >> 5;

    float w20[4], w21[4], syn1[4] = {0,0,0,0}, mem1[4] = {0,0,0,0};
    #pragma unroll
    for (int j = 0; j < 4; ++j) {
        int h = tid + 256 * j;
        w20[j] = w2[h];
        w21[j] = w2[NH + h];
    }
    float syn2a = 0.f, syn2b = 0.f, mem2a = 0.f, mem2b = 0.f;
    const float b2a = b2[0], b2b = b2[1];
    __shared__ float red[2][2][8];

    const float* cb = cur1 + (size_t)b * T * NH;
    float c[4];
    #pragma unroll
    for (int j = 0; j < 4; ++j) c[j] = cb[tid + 256 * j];

    for (int t = 0; t < T; ++t) {
        // software-pipelined load of next timestep
        float cn[4] = {0.f, 0.f, 0.f, 0.f};
        if (t + 1 < T) {
            const float* np = cb + (size_t)(t + 1) * NH;
            __builtin_prefetch(np + tid, 0, 3);
            #pragma unroll
            for (int j = 0; j < 4; ++j) cn[j] = np[tid + 256 * j];
        }

        // layer-1 Synaptic update (4 neurons/thread) + fc2 partials
        float p0 = 0.f, p1 = 0.f;
        #pragma unroll
        for (int j = 0; j < 4; ++j) {
            float r1 = (mem1[j] > THR) ? THR : 0.f;   // reset from pre-update mem
            syn1[j] = ALPHA * syn1[j] + c[j];
            mem1[j] = BETA * mem1[j] + syn1[j] - r1;
            if (mem1[j] - THR > 0.f) { p0 += w20[j]; p1 += w21[j]; }  // spk1 in {0,1}
        }

        // reduce 1024 -> 2 across the workgroup
        p0 = wave_sum(p0);
        p1 = wave_sum(p1);
        if (lane == 0) { red[t & 1][0][wid] = p0; red[t & 1][1][wid] = p1; }
        __syncthreads();

        if (tid == 0) {
            float s0 = b2a, s1 = b2b;
            #pragma unroll
            for (int w = 0; w < 8; ++w) { s0 += red[t & 1][0][w]; s1 += red[t & 1][1][w]; }
            float r2a = (mem2a > THR) ? THR : 0.f;
            float r2b = (mem2b > THR) ? THR : 0.f;
            syn2a = ALPHA * syn2a + s0; mem2a = BETA * mem2a + syn2a - r2a;
            syn2b = ALPHA * syn2b + s1; mem2b = BETA * mem2b + syn2b - r2b;
            float sp2a = (mem2a - THR > 0.f) ? 1.f : 0.f;
            float sp2b = (mem2b - THR > 0.f) ? 1.f : 0.f;
            size_t base = ((size_t)b * T + t) * 2;
            out[base]     = mem2a;
            out[base + 1] = mem2b;
            out[(size_t)B * T * 2 + base]     = sp2a;
            out[(size_t)B * T * 2 + base + 1] = sp2b;
        }
        #pragma unroll
        for (int j = 0; j < 4; ++j) c[j] = cn[j];
    }
}

// ================= host launcher =================
extern "C" void kernel_launch(void* const* d_in, const int* in_sizes, int n_in,
                              void* d_out, int out_size, void* d_ws, size_t ws_size,
                              hipStream_t stream) {
    const float* x  = (const float*)d_in[0];  // [128,200,64,64]
    const float* w1 = (const float*)d_in[1];  // [1024,4096]
    const float* b1 = (const float*)d_in[2];  // [1024]
    const float* w2 = (const float*)d_in[3];  // [2,1024]
    const float* b2 = (const float*)d_in[4];  // [2]
    float* out = (float*)d_out;

    constexpr int B = 128, T = 200, NH = 1024, NI = 4096;
    const size_t M = (size_t)B * T;  // 25600

    // workspace layout: [cur1: M*NH fp32 = 104.9 MB][w1 bf16: NH*NI*2 = 8.4 MB]
    float* cur1 = (float*)d_ws;
    unsigned short* w1bf = (unsigned short*)((char*)d_ws + M * NH * sizeof(float));

    const int n4 = NH * NI / 4;  // float4 chunks of w1
    hipLaunchKernelGGL(cvt_w1_kernel, dim3((n4 + 255) / 256), dim3(256), 0, stream,
                       w1, w1bf, n4);
    hipLaunchKernelGGL(fc1_gemm_kernel, dim3(NH / 128, (int)(M / 128)), dim3(256), 0, stream,
                       x, w1bf, b1, cur1);
    hipLaunchKernelGGL(snn_scan_kernel, dim3(B), dim3(256), 0, stream,
                       cur1, w2, b2, out);
}
